// STFT_32435593019847
// MI455X (gfx1250) — matrix operand order repaired and checked by reading, not compiled, not run on hardware
//
#include <hip/hip_runtime.h>
#include <hip/hip_bf16.h>

typedef __attribute__((ext_vector_type(16))) _Float16 v16h;
typedef __attribute__((ext_vector_type(8)))  _Float16 v8h;
typedef __attribute__((ext_vector_type(8)))  float    v8f;

#define T_LEN   524288
#define NBATCH  32
#define PADL    512
#define NCH     1026            /* 2*513 basis rows */
#define PADDED  (T_LEN + 2*PADL)/* 525312 */
#define KU      1536            /* combined operator K-dim */
#define NJ      512             /* output samples per chunk */
#define MT      32              /* chunks per block in main kernel */
#define SPAN    (MT*512 + 1024) /* 17408 f16 staged per block */
#define NKK     48              /* K steps of 32 */
#define SCALE_UP 1048576.0f
#define SCALE_DN (1.0f/1048576.0f)

// ---------------------------------------------------------------------------
// Kernel 1: G32[j][u] = K[j+512,u]*(u<1024) + K[j,u-512]*(u>=512),
//           K[k,t] = sum_c inv[c,k]*fwd[c,t].  f32, LDS-tiled.
// ---------------------------------------------------------------------------
__global__ __launch_bounds__(256) void k_build_G(const float* __restrict__ fwd,
                                                 const float* __restrict__ inv,
                                                 float* __restrict__ G32) {
    __shared__ float sInvHi[32][16];
    __shared__ float sInvLo[32][16];
    __shared__ float sFwdA[32][64];
    __shared__ float sFwdB[32][64];
    const int j0 = blockIdx.x * 16;   // 32 tiles over NJ
    const int u0 = blockIdx.y * 64;   // 24 tiles over KU
    const int tid = threadIdx.x;
    const int jl = tid >> 4;          // 0..15
    const int us = tid & 15;          // 0..15 -> 4 u's each
    const bool useA = (u0 < 1024);
    const bool useB = (u0 >= 512);
    float acc[4] = {0.f, 0.f, 0.f, 0.f};
    for (int cb = 0; cb < NCH; cb += 32) {
        for (int idx = tid; idx < 512; idx += 256) {
            int cc = idx >> 4, jj = idx & 15;
            int c = cb + cc;
            sInvHi[cc][jj] = (c < NCH) ? inv[c*1024 + j0 + jj + 512] : 0.f;
            sInvLo[cc][jj] = (c < NCH) ? inv[c*1024 + j0 + jj]       : 0.f;
        }
        for (int idx = tid; idx < 2048; idx += 256) {
            int cc = idx >> 6, uu = idx & 63;
            int c = cb + cc;
            sFwdA[cc][uu] = (useA && c < NCH) ? fwd[c*1024 + u0 + uu]       : 0.f;
            sFwdB[cc][uu] = (useB && c < NCH) ? fwd[c*1024 + u0 - 512 + uu] : 0.f;
        }
        __syncthreads();
        #pragma unroll 4
        for (int cc = 0; cc < 32; ++cc) {
            float ih = sInvHi[cc][jl];
            float il = sInvLo[cc][jl];
            #pragma unroll
            for (int q = 0; q < 4; ++q)
                acc[q] += ih * sFwdA[cc][4*us + q] + il * sFwdB[cc][4*us + q];
        }
        __syncthreads();
    }
    #pragma unroll
    for (int q = 0; q < 4; ++q)
        G32[(size_t)(j0 + jl) * KU + u0 + 4*us + q] = acc[q];
}

// ---------------------------------------------------------------------------
// Kernel 2: extract exact diagonal d[j]=G32[j][j+512]; residual*2^20 -> f16
// repacked FRAGMENT-MAJOR:  Gp[ct][kk][lane][16]  (ct=j>>4, kk=u>>5,
// lane=16*((u>>3)&1)+(j&15), h=(u&7)+8*((u>>4)&1))  -> per-wave B loads fully coalesced.
// Also wsinv[j] = 1/(w[j]^2 + w[j+512]^2 + 1e-6).
// ---------------------------------------------------------------------------
__global__ __launch_bounds__(256) void k_prep(const float* __restrict__ G32,
                                              const float* __restrict__ win,
                                              _Float16* __restrict__ Gp,
                                              float* __restrict__ dscale,
                                              float* __restrict__ wsinv) {
    int id = blockIdx.x * 256 + threadIdx.x;
    if (id < NJ * KU) {
        int j = id / KU;
        int u = id - j * KU;
        float v = G32[id];
        if (u == j + 512) { dscale[j] = v; v = 0.f; }
        int ct = j >> 4, n = j & 15;
        int kk = u >> 5, r = (u >> 3) & 1, h = (u & 7) + 8 * ((u >> 4) & 1);
        size_t off = (((size_t)ct * NKK + kk) * 32 + (16 * r + n)) * 16 + h;
        Gp[off] = (_Float16)(v * SCALE_UP);
    }
    if (id < NJ) {
        float w0 = win[id], w1 = win[id + 512];
        wsinv[id] = 1.0f / (w0*w0 + w1*w1 + 1e-6f);
    }
}

// ---------------------------------------------------------------------------
// Kernel 3: reflect-padded f16 copy of the input (B x PADDED)
// ---------------------------------------------------------------------------
__global__ __launch_bounds__(256) void k_pad16(const float* __restrict__ x,
                                               _Float16* __restrict__ xp) {
    long long id = (long long)blockIdx.x * 256 + threadIdx.x;
    if (id >= (long long)NBATCH * PADDED) return;
    int b = (int)(id / PADDED);
    int i = (int)(id - (long long)b * PADDED);
    int idx = i - PADL;
    if (idx < 0) idx = -idx;
    else if (idx >= T_LEN) idx = 2*(T_LEN - 1) - idx;
    xp[id] = (_Float16)x[(size_t)b * T_LEN + idx];
}

// ---------------------------------------------------------------------------
// Kernel 4: out[b, 512c+j] = (WMMA_resid + x*diag) * wsinv
//   8 waves each own a disjoint 64-wide N slice; all share the 32 M-rows.
//   K-loop unrolled x2 with ping-pong B buffers (no register rotation).
// ---------------------------------------------------------------------------
__global__ __launch_bounds__(256) void k_istft_wmma(
        const _Float16* __restrict__ xp,
        const _Float16* __restrict__ Gp,
        const float* __restrict__ dscale,
        const float* __restrict__ wsinv,
        const float* __restrict__ x,
        float* __restrict__ out) {
    __shared__ _Float16 xs[SPAN];
    const int cblk = blockIdx.x;    // 0..31 (chunk block)
    const int b    = blockIdx.y;    // 0..31 (batch)
    const int tid  = threadIdx.x;
    const int lane = tid & 31;
    const int wave = tid >> 5;      // 0..7 -> N base = 64*wave

    // Stage the 17408-sample f16 x-span into LDS with async b128 copies.
    // (wave-uniform trip counts -> EXEC fully set on every issued op)
    {
        const _Float16* gsrc = xp + (size_t)b * PADDED + (size_t)cblk * (MT * 512);
        unsigned ldsbase = (unsigned)(uintptr_t)(&xs[0]);
        for (int i = tid; i < SPAN / 8; i += 256) {
            unsigned loff = ldsbase + 16u * (unsigned)i;
            const _Float16* gsrc8 = gsrc + 8 * i;
            asm volatile("global_load_async_to_lds_b128 %0, %1, off"
                         :: "v"(loff), "v"(gsrc8) : "memory");
        }
        asm volatile("s_wait_asynccnt 0x0" ::: "memory");
    }
    __syncthreads();

    v8f acc[8] = {};                // [4*mt + nt]
    const int m = lane & 15;        // row (A) / col (B) within a 16-tile
    const int g = lane >> 4;        // lane-group

    // Fragment-major B pointers: one contiguous 32B per lane, coalesced/wave
    const _Float16* bbase[4];
    #pragma unroll
    for (int nt = 0; nt < 4; ++nt) {
        int ct = 4 * wave + nt;                       // 16-col tile index
        bbase[nt] = Gp + (size_t)ct * (NKK * 512) + (size_t)lane * 16;
    }

    const int arow0 = 512 * m;            // M-tile 0 row base (halves)
    const int arow1 = 512 * (16 + m);     // M-tile 1 row base

    v16h bA[4], bB[4];
    #pragma unroll
    for (int nt = 0; nt < 4; ++nt)
        bA[nt] = *(const v16h*)(bbase[nt]);           // kk = 0

    for (int kk = 0; kk < NKK; kk += 2) {
        // prefetch B for kk+1 into the other buffer
        #pragma unroll
        for (int nt = 0; nt < 4; ++nt)
            bB[nt] = *(const v16h*)(bbase[nt] + (size_t)(kk + 1) * 512);

        {   // compute with bA at kk
            const int u0 = 32 * kk;
            v8h lo0 = *(const v8h*)(xs + arow0 + u0 + 8 * g);
            v8h hi0 = *(const v8h*)(xs + arow0 + u0 + 8 * g + 16);
            v8h lo1 = *(const v8h*)(xs + arow1 + u0 + 8 * g);
            v8h hi1 = *(const v8h*)(xs + arow1 + u0 + 8 * g + 16);
            v16h a0 = __builtin_shufflevector(lo0, hi0,
                      0,1,2,3,4,5,6,7,8,9,10,11,12,13,14,15);
            v16h a1 = __builtin_shufflevector(lo1, hi1,
                      0,1,2,3,4,5,6,7,8,9,10,11,12,13,14,15);
            #pragma unroll
            for (int nt = 0; nt < 4; ++nt) {
                acc[nt]     = __builtin_amdgcn_wmma_f32_16x16x32_f16(
                                  false, a0, false, bA[nt], (short)0, acc[nt],     false, false);
                acc[4 + nt] = __builtin_amdgcn_wmma_f32_16x16x32_f16(
                                  false, a1, false, bA[nt], (short)0, acc[4 + nt], false, false);
            }
        }

        // prefetch B for kk+2 back into bA
        if (kk + 2 < NKK) {
            #pragma unroll
            for (int nt = 0; nt < 4; ++nt)
                bA[nt] = *(const v16h*)(bbase[nt] + (size_t)(kk + 2) * 512);
        }

        {   // compute with bB at kk+1
            const int u0 = 32 * (kk + 1);
            v8h lo0 = *(const v8h*)(xs + arow0 + u0 + 8 * g);
            v8h hi0 = *(const v8h*)(xs + arow0 + u0 + 8 * g + 16);
            v8h lo1 = *(const v8h*)(xs + arow1 + u0 + 8 * g);
            v8h hi1 = *(const v8h*)(xs + arow1 + u0 + 8 * g + 16);
            v16h a0 = __builtin_shufflevector(lo0, hi0,
                      0,1,2,3,4,5,6,7,8,9,10,11,12,13,14,15);
            v16h a1 = __builtin_shufflevector(lo1, hi1,
                      0,1,2,3,4,5,6,7,8,9,10,11,12,13,14,15);
            #pragma unroll
            for (int nt = 0; nt < 4; ++nt) {
                acc[nt]     = __builtin_amdgcn_wmma_f32_16x16x32_f16(
                                  false, a0, false, bB[nt], (short)0, acc[nt],     false, false);
                acc[4 + nt] = __builtin_amdgcn_wmma_f32_16x16x32_f16(
                                  false, a1, false, bB[nt], (short)0, acc[4 + nt], false, false);
            }
        }
    }

    // Epilogue: add exact-f32 diagonal term, divide by window sum
    #pragma unroll
    for (int nt = 0; nt < 4; ++nt) {
        const int colN = 64 * wave + 16 * nt + m;   // 0..511
        const float dj = dscale[colN];
        const float wj = wsinv[colN];
        #pragma unroll
        for (int mt = 0; mt < 2; ++mt) {
            #pragma unroll
            for (int r = 0; r < 8; ++r) {
                const int chunk = 32 * cblk + 16 * mt + 8 * g + r;  // 0..1023
                const size_t o = (size_t)b * T_LEN + (size_t)chunk * 512 + colN;
                out[o] = (acc[4 * mt + nt][r] * SCALE_DN + x[o] * dj) * wj;
            }
        }
    }
}

// ---------------------------------------------------------------------------
extern "C" void kernel_launch(void* const* d_in, const int* in_sizes, int n_in,
                              void* d_out, int out_size, void* d_ws, size_t ws_size,
                              hipStream_t stream) {
    const float* x    = (const float*)d_in[0];  // (32, 524288)
    const float* fwdb = (const float*)d_in[1];  // (1026, 1024)
    const float* invb = (const float*)d_in[2];  // (1026, 1024)
    const float* win  = (const float*)d_in[3];  // (1024,)
    float* out = (float*)d_out;

    char* ws = (char*)d_ws;
    float*    G32    = (float*)(ws);                               // 3,145,728 B
    _Float16* Gp     = (_Float16*)(ws + 3145728);                  // 1,572,864 B
    float*    dscale = (float*)(ws + 3145728 + 1572864);           // 2 KB
    float*    wsinv  = (float*)(ws + 3145728 + 1572864 + 2048);    // 2 KB
    _Float16* xp     = (_Float16*)(ws + 3145728 + 1572864 + 4096); // 33,619,968 B

    k_build_G<<<dim3(32, 24), 256, 0, stream>>>(fwdb, invb, G32);
    k_prep<<<dim3((NJ * KU + 255) / 256), 256, 0, stream>>>(G32, win, Gp, dscale, wsinv);
    {
        long long tot = (long long)NBATCH * PADDED;
        k_pad16<<<dim3((unsigned)((tot + 255) / 256)), 256, 0, stream>>>(x, xp);
    }
    k_istft_wmma<<<dim3(32, 32), 256, 0, stream>>>(xp, Gp, dscale, wsinv, x, out);
}
